// MultiHeadAttention_38293928411768
// MI455X (gfx1250) — matrix operand order repaired and checked by reading, not compiled, not run on hardware
//
#include <hip/hip_runtime.h>

#define B_  4
#define S_  2048
#define D_  1024
#define H_  16
#define HD_ 64
#define BS_ (B_ * S_)

typedef __bf16 bf16_t;
typedef __attribute__((ext_vector_type(16))) __bf16 v16bf;
typedef __attribute__((ext_vector_type(8)))  __bf16 v8bf;
typedef __attribute__((ext_vector_type(8)))  float  v8f;

static __device__ __forceinline__ v16bf pack16(v8bf lo, v8bf hi) {
  union { v16bf v; v8bf h[2]; } u;
  u.h[0] = lo; u.h[1] = hi;
  return u.v;
}

// A fragment: 16(M) x 32(K) bf16 tile from row-major storage (leading dim ld).
// Lane l (ml=l&15, g=l>>4) holds row M=ml, K = {8g..8g+7} U {16+8g..16+8g+7}.
static __device__ __forceinline__ v16bf load_fragA(const bf16_t* base, int ld,
                                                   int m0, int k0, int lane) {
  const int ml = lane & 15, g = lane >> 4;
  const bf16_t* p = base + (size_t)(m0 + ml) * ld + k0;
  v8bf lo = *(const v8bf*)(p + 8 * g);
  v8bf hi = *(const v8bf*)(p + 16 + 8 * g);
  return pack16(lo, hi);
}

// B fragment: 32(K) x 16(N) bf16 tile from N-major storage (row n holds K
// contiguously, leading dim ld). Lane l (n=l&15, g=l>>4): col N=n, K = {8g..8g+7} U {16+8g..16+8g+7}.
static __device__ __forceinline__ v16bf load_fragB(const bf16_t* base, int ld,
                                                   int n0, int k0, int lane) {
  const int n = lane & 15, g = lane >> 4;
  const bf16_t* p = base + (size_t)(n0 + n) * ld + k0;
  v8bf lo = *(const v8bf*)(p + 8 * g);
  v8bf hi = *(const v8bf*)(p + 16 + 8 * g);
  return pack16(lo, hi);
}

#define WMMA_BF16(A, Bf, C) \
  __builtin_amdgcn_wmma_f32_16x16x32_bf16(false, (A), false, (Bf), (short)0, (C), false, false)

// ---------------------------------------------------------------------------
// Stage 0: fp32 -> bf16 convert / weight transpose to N-major
// ---------------------------------------------------------------------------
__global__ void cvt_bf16_kernel(const float* __restrict__ in,
                                bf16_t* __restrict__ out, int n) {
  int i = blockIdx.x * blockDim.x + threadIdx.x;
  if (i < n) out[i] = (bf16_t)in[i];
}

// in: [nmat, K, N] fp32  ->  out: [nmat, N, K] bf16
__global__ void transpose_w_kernel(const float* __restrict__ in,
                                   bf16_t* __restrict__ out,
                                   int nmat, int K, int N) {
  int i = blockIdx.x * blockDim.x + threadIdx.x;
  int total = nmat * K * N;
  if (i >= total) return;
  int m = i / (K * N);
  int rem = i - m * (K * N);
  int k = rem / N;
  int n = rem - k * N;
  out[(size_t)m * N * K + (size_t)n * K + k] = (bf16_t)in[i];
}

// ---------------------------------------------------------------------------
// Stage 1: fused per-head QKV projection.  One wave computes a 64(S) x 64(HD)
// tile: C = x[64x1024] * W_h[1024x64] + bias.  transpose_out=1 stores [e][s]
// (used for V so the PV B-fragments are contiguous later).
// ---------------------------------------------------------------------------
__global__ void __launch_bounds__(32)
qkv_gemm_kernel(const bf16_t* __restrict__ xb,   // [B*S, D] row-major bf16
                const bf16_t* __restrict__ wT,   // [H, HD, D] N-major bf16
                const float*  __restrict__ bias, // [H, HD]
                bf16_t* __restrict__ out,        // [B*H,S,HD] or [B*H,HD,S]
                int transpose_out) {
  const int lane = threadIdx.x;
  const int m0 = blockIdx.x * 64;       // global row in [0, B*S)
  const int h  = blockIdx.y;
  const int b  = m0 / S_;
  const int s0 = m0 - b * S_;

  const bf16_t* wHead = wT + (size_t)h * HD_ * D_;

  v8f acc[4][4];
#pragma unroll
  for (int mi = 0; mi < 4; ++mi)
#pragma unroll
    for (int nj = 0; nj < 4; ++nj) acc[mi][nj] = (v8f)(0.0f);

  for (int k0 = 0; k0 < D_; k0 += 32) {
    // prefetch next k-panel (one lane per 128B line; prefetch skips LOADcnt)
    if (k0 + 32 < D_) {
      __builtin_prefetch(xb + (size_t)(m0 + lane) * D_ + k0 + 32, 0, 3);
      __builtin_prefetch(xb + (size_t)(m0 + 32 + lane) * D_ + k0 + 32, 0, 3);
      __builtin_prefetch(wHead + (size_t)lane * D_ + k0 + 32, 0, 3);
      __builtin_prefetch(wHead + (size_t)(32 + lane) * D_ + k0 + 32, 0, 3);
    }
    v16bf a[4], bf[4];
#pragma unroll
    for (int mi = 0; mi < 4; ++mi)
      a[mi] = load_fragA(xb, D_, m0 + 16 * mi, k0, lane);
#pragma unroll
    for (int nj = 0; nj < 4; ++nj)
      bf[nj] = load_fragB(wHead, D_, 16 * nj, k0, lane);
#pragma unroll
    for (int mi = 0; mi < 4; ++mi)
#pragma unroll
      for (int nj = 0; nj < 4; ++nj)
        acc[mi][nj] = WMMA_BF16(a[mi], bf[nj], acc[mi][nj]);
  }

  const int n = lane & 15, g = lane >> 4;
#pragma unroll
  for (int mi = 0; mi < 4; ++mi) {
#pragma unroll
    for (int nj = 0; nj < 4; ++nj) {
      const int e = nj * 16 + n;
      const float bv = bias[h * HD_ + e];
#pragma unroll
      for (int r = 0; r < 8; ++r) {
        const int srow = s0 + mi * 16 + r + 8 * g;
        const bf16_t o = (bf16_t)(acc[mi][nj][r] + bv);
        if (!transpose_out)
          out[((size_t)(b * H_ + h) * S_ + srow) * HD_ + e] = o;
        else
          out[((size_t)(b * H_ + h) * HD_ + e) * S_ + srow] = o;
      }
    }
  }
}

// ---------------------------------------------------------------------------
// Stage 2: causal flash attention.  One wave owns 32 query rows of one (b,h);
// Q fragments live in registers for the whole kernel, K/Vt are streamed,
// online softmax (max/sum) kept in fp32 with shfl_xor reductions across the
// 16-lane group that shares an accumulator row set.
// ---------------------------------------------------------------------------
__global__ void __launch_bounds__(32)
flash_attn_kernel(const bf16_t* __restrict__ Qb,  // [B*H, S, HD]
                  const bf16_t* __restrict__ Kb,  // [B*H, S, HD]
                  const bf16_t* __restrict__ Vt,  // [B*H, HD, S]
                  bf16_t* __restrict__ Cb) {      // [B, S, D] concat output
  __shared__ __align__(16) bf16_t Pld[32 * 32];   // P tile, row-major bf16

  const int lane = threadIdx.x;
  const int iblk = blockIdx.x;          // S/32 row blocks
  const int bh   = blockIdx.y;          // B*H
  const int b    = bh / H_;
  const int h    = bh - b * H_;
  const int r0   = iblk * 32;

  const bf16_t* Qp = Qb + (size_t)bh * S_ * HD_;
  const bf16_t* Kp = Kb + (size_t)bh * S_ * HD_;
  const bf16_t* Vp = Vt + (size_t)bh * HD_ * S_;

  const int n = lane & 15, g = lane >> 4;
  const float scale = 0.125f;  // 1/sqrt(64)

  // Preload Q: 2 M-frags x 2 K-steps (HD=64)
  v16bf qf[2][2];
#pragma unroll
  for (int mi = 0; mi < 2; ++mi)
#pragma unroll
    for (int kk = 0; kk < 2; ++kk)
      qf[mi][kk] = load_fragA(Qp, HD_, r0 + 16 * mi, 32 * kk, lane);

  v8f o[2][4];
  float mrow[2][8], lrow[2][8];
#pragma unroll
  for (int mi = 0; mi < 2; ++mi) {
#pragma unroll
    for (int ej = 0; ej < 4; ++ej) o[mi][ej] = (v8f)(0.0f);
#pragma unroll
    for (int r = 0; r < 8; ++r) { mrow[mi][r] = -3.0e38f; lrow[mi][r] = 0.0f; }
  }

  for (int j = 0; j <= iblk; ++j) {
    const int c0 = j * 32;

    // prefetch next iteration's K rows (32 rows x 128B) and Vt rows (64 x 64B)
    if (j < iblk) {
      __builtin_prefetch(Kp + (size_t)(c0 + 32 + lane) * HD_, 0, 3);
      __builtin_prefetch(Vp + (size_t)lane * S_ + c0 + 32, 0, 3);
      __builtin_prefetch(Vp + (size_t)(32 + lane) * S_ + c0 + 32, 0, 3);
    }

    // ---- scores S = Q K^T * scale ----
    v8f sc[2][2];
#pragma unroll
    for (int mi = 0; mi < 2; ++mi)
#pragma unroll
      for (int nj = 0; nj < 2; ++nj) sc[mi][nj] = (v8f)(0.0f);

#pragma unroll
    for (int kk = 0; kk < 2; ++kk) {
      v16bf kf[2];
#pragma unroll
      for (int nj = 0; nj < 2; ++nj)
        kf[nj] = load_fragB(Kp, HD_, c0 + 16 * nj, 32 * kk, lane);
#pragma unroll
      for (int mi = 0; mi < 2; ++mi)
#pragma unroll
        for (int nj = 0; nj < 2; ++nj)
          sc[mi][nj] = WMMA_BF16(qf[mi][kk], kf[nj], sc[mi][nj]);
    }

    // ---- online softmax (per 16-row M-frag; row M = r + 8g in C layout) ----
#pragma unroll
    for (int mi = 0; mi < 2; ++mi) {
#pragma unroll
      for (int nj = 0; nj < 2; ++nj) {
#pragma unroll
        for (int r = 0; r < 8; ++r) {
          float v = sc[mi][nj][r] * scale;
          const int col = c0 + nj * 16 + n;
          const int row = r0 + mi * 16 + r + 8 * g;
          if (col > row) v = -1.0e30f;  // causal mask
          sc[mi][nj][r] = v;
        }
      }
#pragma unroll
      for (int r = 0; r < 8; ++r) {
        float tm = fmaxf(sc[mi][0][r], sc[mi][1][r]);
#pragma unroll
        for (int off = 1; off < 16; off <<= 1)
          tm = fmaxf(tm, __shfl_xor(tm, off));
        const float mold = mrow[mi][r];
        const float mnew = fmaxf(mold, tm);
        const float alpha = __expf(mold - mnew);
        mrow[mi][r] = mnew;
        lrow[mi][r] *= alpha;
#pragma unroll
        for (int ej = 0; ej < 4; ++ej) o[mi][ej][r] *= alpha;
        float psum = 0.0f;
#pragma unroll
        for (int nj = 0; nj < 2; ++nj) {
          const float p = __expf(sc[mi][nj][r] - mnew);
          sc[mi][nj][r] = p;
          psum += p;
        }
#pragma unroll
        for (int off = 1; off < 16; off <<= 1)
          psum += __shfl_xor(psum, off);
        lrow[mi][r] += psum;
      }
    }

    // ---- P (C layout) -> LDS row-major bf16, reload as A fragments ----
#pragma unroll
    for (int mi = 0; mi < 2; ++mi)
#pragma unroll
      for (int nj = 0; nj < 2; ++nj)
#pragma unroll
        for (int r = 0; r < 8; ++r)
          Pld[(mi * 16 + r + 8 * g) * 32 + nj * 16 + n] = (bf16_t)sc[mi][nj][r];

    v16bf pf[2];
#pragma unroll
    for (int mi = 0; mi < 2; ++mi)
      pf[mi] = load_fragA(Pld, 32, 16 * mi, 0, lane);

    // ---- O += P * V  (V^T streamed: rows = e, contiguous t) ----
#pragma unroll
    for (int ej = 0; ej < 4; ++ej) {
      v16bf vfrag = load_fragB(Vp, S_, 16 * ej, c0, lane);
#pragma unroll
      for (int mi = 0; mi < 2; ++mi)
        o[mi][ej] = WMMA_BF16(pf[mi], vfrag, o[mi][ej]);
    }
  }

  // ---- normalize and scatter into concat buffer [b][s][h*64+e] ----
#pragma unroll
  for (int mi = 0; mi < 2; ++mi) {
#pragma unroll
    for (int ej = 0; ej < 4; ++ej) {
#pragma unroll
      for (int r = 0; r < 8; ++r) {
        const float v = o[mi][ej][r] / lrow[mi][r];
        const int srow = r0 + mi * 16 + r + 8 * g;
        const int col = h * HD_ + ej * 16 + n;
        Cb[((size_t)b * S_ + srow) * D_ + col] = (bf16_t)v;
      }
    }
  }
}

// ---------------------------------------------------------------------------
// Stage 3: output projection Y = concat * Wo + bo (64x64 tile per wave, fp32 out)
// ---------------------------------------------------------------------------
__global__ void __launch_bounds__(32)
out_proj_kernel(const bf16_t* __restrict__ Cb,   // [B*S, D] bf16
                const bf16_t* __restrict__ woT,  // [D(n), D(k)] N-major bf16
                const float*  __restrict__ bo,   // [D]
                float* __restrict__ Y) {         // [B*S, D] fp32
  const int lane = threadIdx.x;
  const int m0 = blockIdx.x * 64;
  const int n0 = blockIdx.y * 64;

  v8f acc[4][4];
#pragma unroll
  for (int mi = 0; mi < 4; ++mi)
#pragma unroll
    for (int nj = 0; nj < 4; ++nj) acc[mi][nj] = (v8f)(0.0f);

  for (int k0 = 0; k0 < D_; k0 += 32) {
    if (k0 + 32 < D_) {
      __builtin_prefetch(Cb + (size_t)(m0 + lane) * D_ + k0 + 32, 0, 3);
      __builtin_prefetch(Cb + (size_t)(m0 + 32 + lane) * D_ + k0 + 32, 0, 3);
      __builtin_prefetch(woT + (size_t)(n0 + lane) * D_ + k0 + 32, 0, 3);
      __builtin_prefetch(woT + (size_t)(n0 + 32 + lane) * D_ + k0 + 32, 0, 3);
    }
    v16bf a[4];
#pragma unroll
    for (int mi = 0; mi < 4; ++mi)
      a[mi] = load_fragA(Cb, D_, m0 + 16 * mi, k0, lane);
#pragma unroll
    for (int nj = 0; nj < 4; ++nj) {
      v16bf bfrag = load_fragB(woT, D_, n0 + 16 * nj, k0, lane);
#pragma unroll
      for (int mi = 0; mi < 4; ++mi)
        acc[mi][nj] = WMMA_BF16(a[mi], bfrag, acc[mi][nj]);
    }
  }

  const int n = lane & 15, g = lane >> 4;
#pragma unroll
  for (int mi = 0; mi < 4; ++mi) {
#pragma unroll
    for (int nj = 0; nj < 4; ++nj) {
      const float bias = bo[n0 + nj * 16 + n];
#pragma unroll
      for (int r = 0; r < 8; ++r)
        Y[(size_t)(m0 + mi * 16 + r + 8 * g) * D_ + n0 + nj * 16 + n] =
            acc[mi][nj][r] + bias;
    }
  }
}

// ---------------------------------------------------------------------------
extern "C" void kernel_launch(void* const* d_in, const int* in_sizes, int n_in,
                              void* d_out, int out_size, void* d_ws, size_t ws_size,
                              hipStream_t stream) {
  (void)in_sizes; (void)n_in; (void)out_size; (void)ws_size;
  const float* x  = (const float*)d_in[0];
  // d_in[1] = mask (causal, computed analytically in-kernel)
  const float* Wq = (const float*)d_in[2];
  const float* bq = (const float*)d_in[3];
  const float* Wk = (const float*)d_in[4];
  const float* bk = (const float*)d_in[5];
  const float* Wv = (const float*)d_in[6];
  const float* bv = (const float*)d_in[7];
  const float* Wo = (const float*)d_in[8];
  const float* bo = (const float*)d_in[9];
  float* Y = (float*)d_out;

  char* ws = (char*)d_ws;
  size_t off = 0;
  auto walloc = [&](size_t bytes) -> void* {
    void* p = ws + off;
    off = (off + bytes + 255) & ~(size_t)255;
    return p;
  };
  bf16_t* xb   = (bf16_t*)walloc((size_t)BS_ * D_ * 2);        // x bf16
  bf16_t* wqT  = (bf16_t*)walloc((size_t)H_ * HD_ * D_ * 2);   // Wq^T
  bf16_t* wkT  = (bf16_t*)walloc((size_t)H_ * HD_ * D_ * 2);   // Wk^T
  bf16_t* wvT  = (bf16_t*)walloc((size_t)H_ * HD_ * D_ * 2);   // Wv^T
  bf16_t* woT  = (bf16_t*)walloc((size_t)D_ * D_ * 2);         // Wo^T
  bf16_t* Qb   = (bf16_t*)walloc((size_t)B_ * H_ * S_ * HD_ * 2);
  bf16_t* Kb   = (bf16_t*)walloc((size_t)B_ * H_ * S_ * HD_ * 2);
  bf16_t* Vtb  = (bf16_t*)walloc((size_t)B_ * H_ * HD_ * S_ * 2);
  bf16_t* Cb   = (bf16_t*)walloc((size_t)BS_ * D_ * 2);

  // Stage 0: convert + transpose
  {
    int n = BS_ * D_;
    cvt_bf16_kernel<<<(n + 255) / 256, 256, 0, stream>>>(x, xb, n);
    int nw = H_ * D_ * HD_;
    transpose_w_kernel<<<(nw + 255) / 256, 256, 0, stream>>>(Wq, wqT, H_, D_, HD_);
    transpose_w_kernel<<<(nw + 255) / 256, 256, 0, stream>>>(Wk, wkT, H_, D_, HD_);
    transpose_w_kernel<<<(nw + 255) / 256, 256, 0, stream>>>(Wv, wvT, H_, D_, HD_);
    int no = D_ * D_;
    transpose_w_kernel<<<(no + 255) / 256, 256, 0, stream>>>(Wo, woT, 1, D_, D_);
  }

  // Stage 1: Q/K/V projections (V stored transposed [e][s])
  {
    dim3 g(BS_ / 64, H_, 1);
    qkv_gemm_kernel<<<g, 32, 0, stream>>>(xb, wqT, bq, Qb, 0);
    qkv_gemm_kernel<<<g, 32, 0, stream>>>(xb, wkT, bk, Kb, 0);
    qkv_gemm_kernel<<<g, 32, 0, stream>>>(xb, wvT, bv, Vtb, 1);
  }

  // Stage 2: causal flash attention -> concat buffer
  {
    dim3 g(S_ / 32, B_ * H_, 1);
    flash_attn_kernel<<<g, 32, 0, stream>>>(Qb, Kb, Vtb, Cb);
  }

  // Stage 3: output projection
  {
    dim3 g(BS_ / 64, D_ / 64, 1);
    out_proj_kernel<<<g, 32, 0, stream>>>(Cb, woT, bo, Y);
  }
}
